// JoinerGraphModel_11364483465798
// MI455X (gfx1250) — matrix operand order repaired and checked by reading, not compiled, not run on hardware
//
#include <hip/hip_runtime.h>

// ---------------------------------------------------------------------------
// CDNA5 / gfx1250 types for WMMA (wave32)
// ---------------------------------------------------------------------------
typedef __attribute__((ext_vector_type(16))) _Float16 v16h;
typedef __attribute__((ext_vector_type(8)))  float    v8f;

struct __align__(16) U4 { unsigned x, y, z, w; };
struct __align__(16) F4 { float x, y, z, w; };
struct __align__(8)  H4 { _Float16 a, b, c, d; };

union FragAB { v16h h; U4 u[2]; };
union AccU   { v8f v; float f[8]; };

// ---------------------------------------------------------------------------
// WMMA GEMM:  C[M,Nout] = A[M,K] (f16, row-major) @ BT[Nout,K]^T (f16) + bias
// Block = 256 threads = 8 waves. Each wave: one 16-row strip x 64 cols
// (4 x v_wmma_f32_16x16x32_f16 accumulators sharing one A fragment).
// grid = (ceil(M/128), Nout/64). Requires K % 32 == 0, Nout % 64 == 0.
// Boundary rows handled by clamped loads + guarded stores (EXEC stays all-1s
// across WMMA, per ISA 7.12 restriction).
// ---------------------------------------------------------------------------
template <bool HALF_OUT>
__global__ __launch_bounds__(256)
void wmma_gemm_kernel(const _Float16* __restrict__ A,
                      const _Float16* __restrict__ BT,
                      const float* __restrict__ bias,
                      void* __restrict__ Cout,
                      int M, int K, int Nout)
{
    const int lane = threadIdx.x & 31;
    const int wv   = threadIdx.x >> 5;
    const int g    = lane >> 4;   // half-wave group (0 or 1)
    const int nl   = lane & 15;   // row/col index within tile

    const int mTile = blockIdx.x * 128 + wv * 16;
    const int nBase = blockIdx.y * 64;

    // A fragment row (clamped so OOB waves still issue valid loads)
    int m  = mTile + nl;
    int mc = (m < M) ? m : (M - 1);
    const _Float16* arow = A + (size_t)mc * K;

    AccU acc[4] = {};

    for (int k0 = 0; k0 < K; k0 += 32) {
        // A 16x32 f16 fragment (ISA 7.12.2): lane nl holds row nl;
        // halves 0..7  -> K = k0 + g*8 + i
        // halves 8..15 -> K = k0 + 16 + g*8 + (i-8)
        FragAB fa;
        fa.u[0] = *(const U4*)(arow + k0 + g * 8);
        fa.u[1] = *(const U4*)(arow + k0 + 16 + g * 8);

#pragma unroll
        for (int j = 0; j < 4; ++j) {
            // B 32x16 fragment: lane holds column (nl) of B, i.e. row of BT;
            // halves 0..7 -> K = k0 + g*8 + i; halves 8..15 -> K = k0 + 16 + g*8 + (i-8)
            const _Float16* brow =
                BT + (size_t)(nBase + j * 16 + nl) * K + k0 + g * 8;
            FragAB fb;
            fb.u[0] = *(const U4*)(brow);
            fb.u[1] = *(const U4*)(brow + 16);

            acc[j].v = __builtin_amdgcn_wmma_f32_16x16x32_f16(
                false, fa.h, false, fb.h, (short)0, acc[j].v, false, false);
        }
    }

    // C/D layout: VGPR r, lane L -> M = g*8 + r, N = nl
#pragma unroll
    for (int j = 0; j < 4; ++j) {
        const int n    = nBase + j * 16 + nl;
        const float bv = bias ? bias[n] : 0.0f;
#pragma unroll
        for (int r = 0; r < 8; ++r) {
            const int mm = mTile + g * 8 + r;
            if (mm < M) {
                float val = acc[j].f[r] + bv;
                if (HALF_OUT)
                    ((_Float16*)Cout)[(size_t)mm * Nout + n] = (_Float16)val;
                else
                    ((float*)Cout)[(size_t)mm * Nout + n] = val;
            }
        }
    }
}

// ---------------------------------------------------------------------------
// Edge message-passing (the bandwidth-critical stage):
// one wave32 per edge; 32 lanes x float4 = H=128 features.
// e_ij = edge_attr @ We[l] + be[l] computed on the fly from LDS-staged We.
// gate = sigmoid(k[dst] + e + q[src]); atomic scatter of gate*v[src] into agg.
// k/q/v (51 MB each) are L2-resident on the 192 MB L2.
// ---------------------------------------------------------------------------
__global__ __launch_bounds__(256)
void edge_kernel(const int* __restrict__ eidx,      // [2,E]
                 const float* __restrict__ eattr,   // [E,16]
                 const float* __restrict__ We,      // [16,128]
                 const float* __restrict__ be,      // [128]
                 const float* __restrict__ k,       // [N,128]
                 const float* __restrict__ q,       // [N,128]
                 const float* __restrict__ v,       // [N,128]
                 float* __restrict__ agg,           // [N,128]
                 int E)
{
    __shared__ __align__(16) float We_s[16][128];
    __shared__ __align__(16) float be_s[128];
    for (int i = threadIdx.x; i < 16 * 128; i += 256)
        We_s[i >> 7][i & 127] = We[i];
    if (threadIdx.x < 128) be_s[threadIdx.x] = be[threadIdx.x];
    __syncthreads();

    const int lane = threadIdx.x & 31;
    const int wv   = threadIdx.x >> 5;
    const long e   = (long)blockIdx.x * 8 + wv;
    if (e >= E) return;

    const int src = eidx[e];              // j (source)
    const int dst = eidx[(size_t)E + e];  // i (target)
    const int f0  = lane * 4;

    // wave-uniform edge attributes (scalarizable loads)
    float a[16];
#pragma unroll
    for (int d = 0; d < 16; ++d) a[d] = eattr[e * 16 + d];

    F4 ee = *(const F4*)(be_s + f0);
#pragma unroll
    for (int d = 0; d < 16; ++d) {
        F4 w = *(const F4*)(&We_s[d][f0]);
        ee.x += a[d] * w.x; ee.y += a[d] * w.y;
        ee.z += a[d] * w.z; ee.w += a[d] * w.w;
    }

    F4 kd = *(const F4*)(k + (size_t)dst * 128 + f0);
    F4 qs = *(const F4*)(q + (size_t)src * 128 + f0);
    F4 vs = *(const F4*)(v + (size_t)src * 128 + f0);

    float g0 = kd.x + ee.x + qs.x, g1 = kd.y + ee.y + qs.y;
    float g2 = kd.z + ee.z + qs.z, g3 = kd.w + ee.w + qs.w;
    g0 = 1.0f / (1.0f + __expf(-g0));
    g1 = 1.0f / (1.0f + __expf(-g1));
    g2 = 1.0f / (1.0f + __expf(-g2));
    g3 = 1.0f / (1.0f + __expf(-g3));

    float* ap = agg + (size_t)dst * 128 + f0;
    atomicAdd(ap + 0, g0 * vs.x);
    atomicAdd(ap + 1, g1 * vs.y);
    atomicAdd(ap + 2, g2 * vs.z);
    atomicAdd(ap + 3, g3 * vs.w);
}

// ---------------------------------------------------------------------------
// Node update: h = gelu(layernorm(agg + skip)); one wave32 per node,
// wave-level allreduce over H=128 via __shfl_xor (wave32 on gfx1250).
// Writes h directly as f16 for the next layer's WMMA GEMMs.
// ---------------------------------------------------------------------------
__global__ __launch_bounds__(256)
void node_kernel(const float* __restrict__ agg, const float* __restrict__ s,
                 const float* __restrict__ lg, const float* __restrict__ lb,
                 _Float16* __restrict__ h_out, int N)
{
    const int lane = threadIdx.x & 31;
    const int wv   = threadIdx.x >> 5;
    const long nd  = (long)blockIdx.x * 8 + wv;
    if (nd >= N) return;
    const int f0 = lane * 4;

    F4 av = *(const F4*)(agg + nd * 128 + f0);
    F4 sv = *(const F4*)(s + nd * 128 + f0);
    float h0 = av.x + sv.x, h1 = av.y + sv.y, h2 = av.z + sv.z, h3 = av.w + sv.w;

    float ls = h0 + h1 + h2 + h3;
    float lq = h0 * h0 + h1 * h1 + h2 * h2 + h3 * h3;
#pragma unroll
    for (int off = 16; off >= 1; off >>= 1) {
        ls += __shfl_xor(ls, off, 32);
        lq += __shfl_xor(lq, off, 32);
    }
    const float mu  = ls * (1.0f / 128.0f);
    const float var = lq * (1.0f / 128.0f) - mu * mu;
    const float inv = rsqrtf(var + 1e-5f);

    const float kInvSqrt2 = 0.70710678118654752f;
    float y0 = (h0 - mu) * inv * lg[f0 + 0] + lb[f0 + 0];
    float y1 = (h1 - mu) * inv * lg[f0 + 1] + lb[f0 + 1];
    float y2 = (h2 - mu) * inv * lg[f0 + 2] + lb[f0 + 2];
    float y3 = (h3 - mu) * inv * lg[f0 + 3] + lb[f0 + 3];
    y0 = 0.5f * y0 * (1.0f + erff(y0 * kInvSqrt2));
    y1 = 0.5f * y1 * (1.0f + erff(y1 * kInvSqrt2));
    y2 = 0.5f * y2 * (1.0f + erff(y2 * kInvSqrt2));
    y3 = 0.5f * y3 * (1.0f + erff(y3 * kInvSqrt2));

    H4 o; o.a = (_Float16)y0; o.b = (_Float16)y1; o.c = (_Float16)y2; o.d = (_Float16)y3;
    *(H4*)(h_out + nd * 128 + f0) = o;
}

// ---------------------------------------------------------------------------
// Small utility kernels
// ---------------------------------------------------------------------------
__global__ void f32_to_f16_kernel(const float* __restrict__ src,
                                  _Float16* __restrict__ dst, long n)
{
    long i = (long)blockIdx.x * blockDim.x + threadIdx.x;
    if (i < n) dst[i] = (_Float16)src[i];
}

// W[K,Nout] f32 -> Wt[Nout,K] f16  (transposed so B fragments are contiguous)
__global__ void transpose_to_half_kernel(const float* __restrict__ W,
                                         _Float16* __restrict__ Wt,
                                         int K, int Nout)
{
    int i = blockIdx.x * blockDim.x + threadIdx.x;
    if (i < K * Nout) {
        int kk = i / Nout, n = i % Nout;
        Wt[(size_t)n * K + kk] = (_Float16)W[i];
    }
}

__global__ void zero_f32_kernel(float* __restrict__ p, long n)
{
    long i = (long)blockIdx.x * blockDim.x + threadIdx.x;
    if (i < n) p[i] = 0.0f;
}

// ---------------------------------------------------------------------------
// Host-side pipeline
// ---------------------------------------------------------------------------
extern "C" void kernel_launch(void* const* d_in, const int* in_sizes, int n_in,
                              void* d_out, int out_size, void* d_ws, size_t ws_size,
                              hipStream_t stream)
{
    (void)n_in; (void)out_size; (void)ws_size;
    const int H = 128, FIN = 256, EDIM = 16, OUT = 64, NL = 3;
    const int N = in_sizes[0] / FIN;
    const int E = in_sizes[1] / 2;

    const float* x     = (const float*)d_in[0];
    const int*   eidx  = (const int*)d_in[1];
    const float* eattr = (const float*)d_in[2];
    const float* in_w  = (const float*)d_in[3];
    const float* in_b  = (const float*)d_in[4];
    const float* Wk    = (const float*)d_in[5];
    const float* bk    = (const float*)d_in[6];
    const float* Wq    = (const float*)d_in[7];
    const float* bq    = (const float*)d_in[8];
    const float* Wv    = (const float*)d_in[9];
    const float* bv    = (const float*)d_in[10];
    const float* We    = (const float*)d_in[11];
    const float* be    = (const float*)d_in[12];
    const float* Ws    = (const float*)d_in[13];
    const float* bs    = (const float*)d_in[14];
    const float* ln_g  = (const float*)d_in[15];
    const float* ln_b  = (const float*)d_in[16];
    const float* out_w = (const float*)d_in[17];
    float* out = (float*)d_out;

    // carve workspace (256B aligned chunks)
    char* ws = (char*)d_ws;
    size_t off = 0;
    auto carve = [&](size_t bytes) -> void* {
        void* p = ws + off;
        off += (bytes + 255) & ~(size_t)255;
        return p;
    };
    _Float16* x_h    = (_Float16*)carve((size_t)N * FIN * 2);
    _Float16* h_h    = (_Float16*)carve((size_t)N * H * 2);
    float*    kbuf   = (float*)carve((size_t)N * H * 4);
    float*    qbuf   = (float*)carve((size_t)N * H * 4);
    float*    vbuf   = (float*)carve((size_t)N * H * 4);
    float*    sbuf   = (float*)carve((size_t)N * H * 4);
    float*    agg    = (float*)carve((size_t)N * H * 4);
    _Float16* in_wT  = (_Float16*)carve((size_t)FIN * H * 2);
    _Float16* WkT    = (_Float16*)carve((size_t)NL * H * H * 2);
    _Float16* WqT    = (_Float16*)carve((size_t)NL * H * H * 2);
    _Float16* WvT    = (_Float16*)carve((size_t)NL * H * H * 2);
    _Float16* WsT    = (_Float16*)carve((size_t)NL * H * H * 2);
    _Float16* out_wT = (_Float16*)carve((size_t)H * OUT * 2);

    // --- weight / input conversion to f16 (transposed weights) ---
    {
        long nx = (long)N * FIN;
        f32_to_f16_kernel<<<(unsigned)((nx + 255) / 256), 256, 0, stream>>>(x, x_h, nx);
        transpose_to_half_kernel<<<(FIN * H + 255) / 256, 256, 0, stream>>>(in_w, in_wT, FIN, H);
        for (int l = 0; l < NL; ++l) {
            const int blks = (H * H + 255) / 256;
            transpose_to_half_kernel<<<blks, 256, 0, stream>>>(Wk + (size_t)l * H * H, WkT + (size_t)l * H * H, H, H);
            transpose_to_half_kernel<<<blks, 256, 0, stream>>>(Wq + (size_t)l * H * H, WqT + (size_t)l * H * H, H, H);
            transpose_to_half_kernel<<<blks, 256, 0, stream>>>(Wv + (size_t)l * H * H, WvT + (size_t)l * H * H, H, H);
            transpose_to_half_kernel<<<blks, 256, 0, stream>>>(Ws + (size_t)l * H * H, WsT + (size_t)l * H * H, H, H);
        }
        transpose_to_half_kernel<<<(H * OUT + 255) / 256, 256, 0, stream>>>(out_w, out_wT, H, OUT);
    }

    const dim3 gemmBlk(256);
    const unsigned gx = (unsigned)((N + 127) / 128);

    // --- in_proj: h = x @ in_w + in_b, stored f16 ---
    wmma_gemm_kernel<true><<<dim3(gx, H / 64), gemmBlk, 0, stream>>>(
        x_h, in_wT, in_b, h_h, N, FIN, H);

    // --- layers ---
    for (int l = 0; l < NL; ++l) {
        wmma_gemm_kernel<false><<<dim3(gx, H / 64), gemmBlk, 0, stream>>>(
            h_h, WkT + (size_t)l * H * H, bk + (size_t)l * H, kbuf, N, H, H);
        wmma_gemm_kernel<false><<<dim3(gx, H / 64), gemmBlk, 0, stream>>>(
            h_h, WqT + (size_t)l * H * H, bq + (size_t)l * H, qbuf, N, H, H);
        wmma_gemm_kernel<false><<<dim3(gx, H / 64), gemmBlk, 0, stream>>>(
            h_h, WvT + (size_t)l * H * H, bv + (size_t)l * H, vbuf, N, H, H);
        wmma_gemm_kernel<false><<<dim3(gx, H / 64), gemmBlk, 0, stream>>>(
            h_h, WsT + (size_t)l * H * H, bs + (size_t)l * H, sbuf, N, H, H);

        long nAgg = (long)N * H;
        zero_f32_kernel<<<(unsigned)((nAgg + 255) / 256), 256, 0, stream>>>(agg, nAgg);

        edge_kernel<<<(unsigned)((E + 7) / 8), 256, 0, stream>>>(
            eidx, eattr, We + (size_t)l * EDIM * H, be + (size_t)l * H,
            kbuf, qbuf, vbuf, agg, E);

        node_kernel<<<(unsigned)((N + 7) / 8), 256, 0, stream>>>(
            agg, sbuf, ln_g + (size_t)l * H, ln_b + (size_t)l * H, h_h, N);
    }

    // --- out_proj: out = h @ out_w (no bias) ---
    wmma_gemm_kernel<false><<<dim3(gx, OUT / 64), gemmBlk, 0, stream>>>(
        h_h, out_wT, (const float*)nullptr, out, N, H, OUT);
}
